// Biaffine_58531814310122
// MI455X (gfx1250) — matrix operand order repaired and checked by reading, not compiled, not run on hardware
//
#include <hip/hip_runtime.h>

typedef __attribute__((ext_vector_type(16))) _Float16 v16h;
typedef __attribute__((ext_vector_type(8)))  _Float16 v8h;
typedef __attribute__((ext_vector_type(8)))  float    v8f;
typedef __attribute__((ext_vector_type(4)))  int      v4i;
// pointer-to-AS1/AS3 v4i (attribute applies to pointee)
typedef __attribute__((address_space(1))) v4i* as1_v4i_p;
typedef __attribute__((address_space(3))) v4i* as3_v4i_p;

#define B_    32
#define S_    1024
#define D_    256
#define O_    2
#define NROWS (B_ * S_)                    // 32768 flattened rows
#define AH_PLANE ((size_t)NROWS * D_)      // elems per o-plane of A
#define LDS_STRIDE 264                     // halves per LDS row (256 + 8 pad -> 132 dwords == 4 mod 64)

#ifndef __has_builtin
#define __has_builtin(x) 0
#endif
#if __has_builtin(__builtin_amdgcn_global_load_async_to_lds_b128) && \
    __has_builtin(__builtin_amdgcn_s_wait_asynccnt)
#define USE_ASYNC_LDS 1
#else
#define USE_ASYNC_LDS 0
#endif

// ---- WMMA wave32 fragment loaders (per CDNA5 ISA layouts) ----
// A (16x32 f16): lane L -> row M=L%16; elems 0..7 = K in [8*(L/16),+8),
// elems 8..15 = K in [16+8*(L/16),+8).
__device__ __forceinline__ v16h load_a_frag(const _Float16* __restrict__ p, int lane) {
  int row  = lane & 15;
  int ksel = (lane >> 4) << 3;
  const _Float16* base = p + row * D_ + ksel;
  v8h lo = *(const v8h*)(base);
  v8h hi = *(const v8h*)(base + 16);
  return __builtin_shufflevector(lo, hi, 0,1,2,3,4,5,6,7,8,9,10,11,12,13,14,15);
}
// B (32x16 f16): lane L -> col N=L%16; elems 0..7 = K in [8*(L/16),+8),
// elems 8..15 = K in [16+8*(L/16),+8).
__device__ __forceinline__ v16h load_b_frag(const _Float16* __restrict__ p, int lane) {
  int col  = lane & 15;
  int koff = (lane >> 4) << 3;
  const _Float16* base = p + col * D_ + koff;
  v8h lo = *(const v8h*)(base);
  v8h hi = *(const v8h*)(base + 16);
  return __builtin_shufflevector(lo, hi, 0,1,2,3,4,5,6,7,8,9,10,11,12,13,14,15);
}

// Same fragments but sourced from LDS (padded stride, 16B-aligned v8h pairs)
__device__ __forceinline__ v16h lds_a_frag(const _Float16* p, int lane, int kb) {
  int row  = lane & 15;
  int ksel = (lane >> 4) << 3;
  const _Float16* base = p + row * LDS_STRIDE + kb + ksel;
  v8h lo = *(const v8h*)(base);
  v8h hi = *(const v8h*)(base + 16);
  return __builtin_shufflevector(lo, hi, 0,1,2,3,4,5,6,7,8,9,10,11,12,13,14,15);
}
__device__ __forceinline__ v16h lds_b_frag(const _Float16* p, int lane, int kb) {
  int col  = lane & 15;
  int koff = (lane >> 4) << 3;
  const _Float16* base = p + col * LDS_STRIDE + kb + koff;
  v8h lo = *(const v8h*)(base);
  v8h hi = *(const v8h*)(base + 16);
  return __builtin_shufflevector(lo, hi, 0,1,2,3,4,5,6,7,8,9,10,11,12,13,14,15);
}

// ---- Kernel 1: f32->f16 convert of input1/input2 + lin1/lin2 dot products ----
__global__ void prep_convert_lin(const float* __restrict__ in1,
                                 const float* __restrict__ in2,
                                 const float* __restrict__ w2,
                                 _Float16* __restrict__ in1_h,
                                 _Float16* __restrict__ in2_h,
                                 float* __restrict__ lin1,
                                 float* __restrict__ lin2) {
  int wid  = blockIdx.x * (blockDim.x >> 5) + (threadIdx.x >> 5);
  int lane = threadIdx.x & 31;
  bool second = wid >= NROWS;
  int row = second ? (wid - NROWS) : wid;
  const float* src = (second ? in2 : in1) + (size_t)row * D_;
  _Float16*    dst = (second ? in2_h : in1_h) + (size_t)row * D_;
  const float* wcol = w2 + (second ? D_ * O_ : 0);

  int j0 = lane * 8;
  float4 f0 = *(const float4*)(src + j0);
  float4 f1 = *(const float4*)(src + j0 + 4);
  v8h h;
  h[0]=(_Float16)f0.x; h[1]=(_Float16)f0.y; h[2]=(_Float16)f0.z; h[3]=(_Float16)f0.w;
  h[4]=(_Float16)f1.x; h[5]=(_Float16)f1.y; h[6]=(_Float16)f1.z; h[7]=(_Float16)f1.w;
  *(v8h*)(dst + j0) = h;

  float vals[8] = {f0.x, f0.y, f0.z, f0.w, f1.x, f1.y, f1.z, f1.w};
  float s0 = 0.f, s1 = 0.f;
#pragma unroll
  for (int k = 0; k < 8; ++k) {
    float2 wv = *(const float2*)(wcol + (size_t)(j0 + k) * O_);
    s0 += vals[k] * wv.x;
    s1 += vals[k] * wv.y;
  }
#pragma unroll
  for (int off = 16; off > 0; off >>= 1) {
    s0 += __shfl_xor(s0, off, 32);
    s1 += __shfl_xor(s1, off, 32);
  }
  if (lane == 0) {
    float* l = (second ? lin2 : lin1) + (size_t)row * O_;
    l[0] = s0; l[1] = s1;
  }
}

// ---- Kernel 2: w1 (D,O,D) f32 -> Wt[c][i] f16, c = o*D + j ----
__global__ void convert_w(const float* __restrict__ w1, _Float16* __restrict__ wt) {
  int c = blockIdx.x;
  int i = threadIdx.x;
  wt[(size_t)c * D_ + i] = (_Float16)w1[(size_t)i * (O_ * D_) + c];
}

// ---- Kernel 3: A[x, c] = in1[x,:] @ W[:,c]  (32768 x 512, K=256), WMMA ----
__global__ void gemm_a(const _Float16* __restrict__ in1_h,
                       const _Float16* __restrict__ wt,
                       _Float16* __restrict__ a_h) {
  int wid  = blockIdx.x * 8 + (threadIdx.x >> 5);
  int lane = threadIdx.x & 31;
  int xt = wid >> 4;
  int ct = wid & 15;
  int x0 = xt * 32, c0 = ct * 32;

  v8f acc[2][2] = {};
  for (int kb = 0; kb < D_; kb += 32) {
    v16h a0 = load_a_frag(in1_h + (size_t)(x0     ) * D_ + kb, lane);
    v16h a1 = load_a_frag(in1_h + (size_t)(x0 + 16) * D_ + kb, lane);
    v16h b0 = load_b_frag(wt    + (size_t)(c0     ) * D_ + kb, lane);
    v16h b1 = load_b_frag(wt    + (size_t)(c0 + 16) * D_ + kb, lane);
    acc[0][0] = __builtin_amdgcn_wmma_f32_16x16x32_f16(false, a0, false, b0, (short)0, acc[0][0], false, false);
    acc[0][1] = __builtin_amdgcn_wmma_f32_16x16x32_f16(false, a0, false, b1, (short)0, acc[0][1], false, false);
    acc[1][0] = __builtin_amdgcn_wmma_f32_16x16x32_f16(false, a1, false, b0, (short)0, acc[1][0], false, false);
    acc[1][1] = __builtin_amdgcn_wmma_f32_16x16x32_f16(false, a1, false, b1, (short)0, acc[1][1], false, false);
  }
#pragma unroll
  for (int mi = 0; mi < 2; ++mi) {
#pragma unroll
    for (int ni = 0; ni < 2; ++ni) {
      int c = c0 + ni * 16 + (lane & 15);
      int o = c >> 8, j = c & 255;
      _Float16* outp = a_h + (size_t)o * AH_PLANE + j;
      int mbase = x0 + mi * 16 + ((lane >> 4) << 3);
#pragma unroll
      for (int r = 0; r < 8; ++r)
        outp[(size_t)(mbase + r) * D_] = (_Float16)acc[mi][ni][r];
    }
  }
}

// ---- Kernel 4: block = 64x128 tile (both o). Stage A+B in LDS via async
// global->LDS DMA, waves consume fragments from LDS, NT b64 output stores. ----
__global__ void __launch_bounds__(256, 2)
biaffine_main(const _Float16* __restrict__ a_h,
              const _Float16* __restrict__ in2_h,
              const float* __restrict__ lin1,
              const float* __restrict__ lin2,
              const float* __restrict__ w2,
              float* __restrict__ out) {
  extern __shared__ char smem_raw[];
  _Float16* smem = (_Float16*)smem_raw;           // A: rows [0,128) (o*64+r), B: rows [128,256)

  int tid  = threadIdx.x;
  int lane = tid & 31;
  int wave = tid >> 5;
  int wx = wave >> 2;            // 0..1 (x sub-tile)
  int wy = wave & 3;             // 0..3 (y sub-tile)

  int bid = blockIdx.x;
  int b   = bid >> 7;            // 128 blocks per batch
  int rem = bid & 127;
  int x0b = (rem >> 3) * 64;     // 16 x-tiles of 64
  int y0b = (rem & 7) * 128;     // 8 y-tiles of 128
  size_t bS = (size_t)b * S_;

  // ---- cooperative fill: thread t copies one 512B row (32 x b128) ----
  const _Float16* gsrc;
  if (tid < 128) {
    int o = tid >> 6, ar = tid & 63;
    gsrc = a_h + (size_t)o * AH_PLANE + (bS + x0b + ar) * D_;
  } else {
    gsrc = in2_h + (bS + y0b + (tid - 128)) * D_;
  }
  _Float16* ldst = smem + (size_t)tid * LDS_STRIDE;

#if USE_ASYNC_LDS
#pragma unroll
  for (int i = 0; i < 32; ++i) {
    __builtin_amdgcn_global_load_async_to_lds_b128(
        (as1_v4i_p)(gsrc + i * 8),
        (as3_v4i_p)(ldst + i * 8),
        0, 0);
  }
  __builtin_amdgcn_s_wait_asynccnt(0);
  __syncthreads();
#else
#pragma unroll
  for (int i = 0; i < 32; ++i) {
    uint4 t = *(const uint4*)(gsrc + i * 8);
    *(uint4*)(ldst + i * 8) = t;
  }
  __syncthreads();
#endif

  const _Float16* aBase = smem;                       // + (o*64 + row)*LDS_STRIDE
  const _Float16* bBase = smem + 128 * LDS_STRIDE;    // + row*LDS_STRIDE

  v8f acc[2][2][2] = {};               // [o][mi][ni]
  for (int kb = 0; kb < D_; kb += 32) {
    v16h a[2][2];
#pragma unroll
    for (int o = 0; o < 2; ++o)
#pragma unroll
      for (int mi = 0; mi < 2; ++mi)
        a[o][mi] = lds_a_frag(aBase + (o * 64 + wx * 32 + mi * 16) * LDS_STRIDE, lane, kb);
    v16h bb[2];
#pragma unroll
    for (int ni = 0; ni < 2; ++ni)
      bb[ni] = lds_b_frag(bBase + (wy * 32 + ni * 16) * LDS_STRIDE, lane, kb);
#pragma unroll
    for (int o = 0; o < 2; ++o)
#pragma unroll
      for (int mi = 0; mi < 2; ++mi)
#pragma unroll
        for (int ni = 0; ni < 2; ++ni)
          acc[o][mi][ni] = __builtin_amdgcn_wmma_f32_16x16x32_f16(
              false, a[o][mi], false, bb[ni], (short)0, acc[o][mi][ni], false, false);
  }

  float bias0 = w2[2 * D_ * O_ + 0];
  float bias1 = w2[2 * D_ * O_ + 1];
#pragma unroll
  for (int mi = 0; mi < 2; ++mi) {
#pragma unroll
    for (int ni = 0; ni < 2; ++ni) {
      int y = y0b + wy * 32 + ni * 16 + (lane & 15);
      float2 l2 = *(const float2*)(lin2 + (bS + y) * O_);
      int mbase = x0b + wx * 32 + mi * 16 + ((lane >> 4) << 3);
#pragma unroll
      for (int r = 0; r < 8; ++r) {
        int m = mbase + r;
        float2 l1 = *(const float2*)(lin1 + (bS + m) * O_);
        union { float f[2]; unsigned long long u; } cv;
        cv.f[0] = acc[0][mi][ni][r] + l1.x + l2.x + bias0;
        cv.f[1] = acc[1][mi][ni][r] + l1.y + l2.y + bias1;
        // non-temporal: 256MB write-once stream must not evict L2 working set
        __builtin_nontemporal_store(cv.u,
            (unsigned long long*)(out + ((bS + m) * S_ + y) * O_));
      }
    }
  }
}

extern "C" void kernel_launch(void* const* d_in, const int* in_sizes, int n_in,
                              void* d_out, int out_size, void* d_ws, size_t ws_size,
                              hipStream_t stream) {
  const float* in1 = (const float*)d_in[0];   // (32,1024,256) f32
  const float* in2 = (const float*)d_in[1];   // (32,1024,256) f32
  const float* w1  = (const float*)d_in[2];   // (256,2,256)   f32
  const float* w2  = (const float*)d_in[3];   // (513,2)       f32
  float* out = (float*)d_out;                 // (32,1024,1024,2) f32
  char* ws = (char*)d_ws;

  _Float16* in1_h = (_Float16*)(ws + 0);          // 16 MB
  _Float16* in2_h = (_Float16*)(ws + 16777216);   // 16 MB
  _Float16* wt    = (_Float16*)(ws + 33554432);   // 256 KB  Wt[c][i]
  _Float16* a_h   = (_Float16*)(ws + 33816576);   // 32 MB   A[o][x][j]
  float*    lin1  = (float*)   (ws + 67371008);   // 256 KB
  float*    lin2  = (float*)   (ws + 67633152);   // 256 KB

  prep_convert_lin<<<8192, 256, 0, stream>>>(in1, in2, w2, in1_h, in2_h, lin1, lin2);
  convert_w<<<512, 256, 0, stream>>>(w1, wt);
  gemm_a<<<2048, 256, 0, stream>>>(in1_h, wt, a_h);
  // dynamic LDS: 256 rows * 264 halves * 2B = 135168 B (fits 320KB WGP budget)
  biaffine_main<<<4096, 256, 135168, stream>>>(a_h, in2_h, lin1, lin2, w2, out);
}
